// GPNStarColAttention_5789615915685
// MI455X (gfx1250) — matrix operand order repaired and checked by reading, not compiled, not run on hardware
//
#include <hip/hip_runtime.h>
#include <stdint.h>

// ---------------------------------------------------------------------------
// GPN Star column attention for MI455X (gfx1250, wave32, WMMA).
// Heavy GEMMs: v_wmma_f32_16x16x32_f16 with weight tiles staged via the CDNA5
// async Global->LDS engine (global_load_async_to_lds_b128 / s_wait_asynccnt),
// converted once to bank-padded f16, then fed to WMMA with ds_load_b128.
// Attention QK^T and PV also on WMMA, fully LDS-fed.
// ---------------------------------------------------------------------------

typedef __attribute__((ext_vector_type(16))) _Float16 v16h;
typedef __attribute__((ext_vector_type(8)))  _Float16 v8h;
typedef __attribute__((ext_vector_type(8)))  float    v8f;

#define NUM_HEADS 6
#define HEAD_DIM  64
#define ALL_HEAD  384
#define HIDDEN    768
#define SEQ_M     65536   // L*S = 1024*64

__device__ __forceinline__ v8f wmma_f16(v16h a, v16h b, v8f c) {
  // (neg_a, A, neg_b, B, c_mod, C, reuse_a, reuse_b)
  return __builtin_amdgcn_wmma_f32_16x16x32_f16(
      false, a, false, b, (short)0, c, false, false);
}

// Async copy of 16 bytes Global -> LDS (per-lane addresses), ASYNCcnt-tracked.
__device__ __forceinline__ void async_g2l_b128(const float* gptr, float* lptr) {
  unsigned lds_off = (unsigned)(uintptr_t)lptr;            // LDS byte offset
  unsigned long long ga = (unsigned long long)(uintptr_t)gptr;
  asm volatile("global_load_async_to_lds_b128 %0, %1, off"
               :: "v"(lds_off), "v"(ga) : "memory");
}
__device__ __forceinline__ void wait_asynccnt0() {
  asm volatile("s_wait_asynccnt 0x0" ::: "memory");
}

// A fragment (16x32 f16 MxK) from GLOBAL f32, row-major, leading dim ld.
// Lane l: row M = l&15; lanes 0-15 K = k0+{0..7,16..23}; lanes 16-31 +8.
__device__ __forceinline__ v16h gmem_a_frag(const float* __restrict__ base,
                                            int ld, int row, int k0, bool hi) {
  const float* p = base + (size_t)row * ld + k0 + (hi ? 8 : 0);
  float4 x0 = *(const float4*)(p);
  float4 x1 = *(const float4*)(p + 4);
  float4 x2 = *(const float4*)(p + 16);
  float4 x3 = *(const float4*)(p + 20);
  v16h a;
  a[0]  = (_Float16)x0.x; a[1]  = (_Float16)x0.y;
  a[2]  = (_Float16)x0.z; a[3]  = (_Float16)x0.w;
  a[4]  = (_Float16)x1.x; a[5]  = (_Float16)x1.y;
  a[6]  = (_Float16)x1.z; a[7]  = (_Float16)x1.w;
  a[8]  = (_Float16)x2.x; a[9]  = (_Float16)x2.y;
  a[10] = (_Float16)x2.z; a[11] = (_Float16)x2.w;
  a[12] = (_Float16)x3.x; a[13] = (_Float16)x3.y;
  a[14] = (_Float16)x3.z; a[15] = (_Float16)x3.w;
  return a;
}

// A fragment from LDS f16 (rowbase = &lds[row*stride]): two aligned v8h loads.
__device__ __forceinline__ v16h lds_a_frag(const _Float16* rowbase, int k0, bool hi) {
  const _Float16* p = rowbase + k0 + (hi ? 8 : 0);
  v8h lo  = *(const v8h*)(p);
  v8h hi8 = *(const v8h*)(p + 16);
  return __builtin_shufflevector(lo, hi8, 0,1,2,3,4,5,6,7,8,9,10,11,12,13,14,15);
}

// B fragment (32x16 f16 KxN) from LDS f16 stored K-contiguous per column
// (colbase = &lds[col*stride]). Lane l: col N = l&15;
// lanes 0-15 K = k0+{0..7,16..23}; lanes 16-31 +8.
__device__ __forceinline__ v16h lds_b_frag(const _Float16* colbase, int k0, bool hi) {
  const _Float16* p = colbase + k0 + (hi ? 8 : 0);
  v8h b0 = *(const v8h*)(p);
  v8h b1 = *(const v8h*)(p + 16);
  return __builtin_shufflevector(b0, b1, 0,1,2,3,4,5,6,7,8,9,10,11,12,13,14,15);
}

// ---------------------------------------------------------------------------
// Kernel 1: combined FIRE bias + attention mask -> bias[64*64]
// ---------------------------------------------------------------------------
__global__ void gpn_fire_bias(const float* __restrict__ phylo,
                              const float* __restrict__ cptr,
                              const float* __restrict__ w1,
                              const float* __restrict__ w2,
                              const float* __restrict__ amask,
                              float* __restrict__ bias) {
  int idx = blockIdx.x * blockDim.x + threadIdx.x;
  if (idx >= 64 * 64) return;
  int kcol = idx & 63;
  float c = fmaxf(cptr[0], 0.0f);
  float d = phylo[idx];
  float r = logf(c * d + 1.0f) / logf(c * 20.0f + 1.0f);
  float acc = 0.0f;
#pragma unroll
  for (int f = 0; f < 32; ++f) {
    float h = r * w1[f];
    float s = h / (1.0f + expf(-h));   // silu
    acc += s * w2[f];
  }
  bias[idx] = acc + amask[kcol];
}

// ---------------------------------------------------------------------------
// Kernel 2: Y[M,N] = X[M,KD] @ W[N,KD]^T + b[N] (+ residual), KD/N/RES static.
// block = 256 threads = 8 waves; wave w -> D tile (blockIdx.y*8+w, blockIdx.x).
// W tile (16 x KD f32) lands in LDS via async Global->LDS DMA, is converted
// once to bank-padded f16, then reused by all 8 waves x KD/32 k-steps via
// ds_load_b128 pairs.
// ---------------------------------------------------------------------------
template<int KD, int NN, bool RES>
__global__ __launch_bounds__(256)
void gpn_gemm_wmma(const float* __restrict__ X, const float* __restrict__ W,
                   const float* __restrict__ bvec,
                   const float* __restrict__ residual,
                   float* __restrict__ Y) {
  constexpr int KF = KD + 4;              // f32 landing stride (dwords)
  constexpr int KP = KD + 8;              // f16 feed stride: 16 lanes -> 16 banks
  __shared__ float    sWf32[16 * KF];     // async landing buffer
  __shared__ _Float16 sW[16 * KP];        // WMMA B feed

  const int ntile = blockIdx.x;

  // ---- async-stage 16 x KD f32 weight tile (b128 per lane) ----
  constexpr int NV = 16 * KD / 4;         // 16B transfers in tile (mult of 256)
  for (int i = threadIdx.x; i < NV; i += 256) {
    int col = i / (KD / 4);
    int k4  = (i % (KD / 4)) * 4;
    async_g2l_b128(W + (size_t)(ntile * 16 + col) * KD + k4,
                   sWf32 + col * KF + k4);
  }
  wait_asynccnt0();                       // this wave's DMAs landed
  __syncthreads();                        // ... and everyone else's

  // ---- one-shot LDS f32 -> padded LDS f16 conversion ----
  for (int i = threadIdx.x; i < 16 * KD; i += 256) {
    int col = i / KD, k = i % KD;
    sW[col * KP + k] = (_Float16)sWf32[col * KF + k];
  }
  __syncthreads();

  const int lane = threadIdx.x & 31;
  const int wave = threadIdx.x >> 5;
  const bool hi  = lane >= 16;
  const int ln   = lane & 15;
  const int mtile = blockIdx.y * 8 + wave;
  const int mrow  = mtile * 16 + ln;
  const int ncol  = ntile * 16 + ln;
  const _Float16* wcol = sW + ln * KP;
  const float* arow = X + (size_t)mrow * KD;

  v8f acc = {};
  for (int k0 = 0; k0 < KD; k0 += 32) {
    __builtin_prefetch(arow + k0 + 64, 0, 0);   // global_prefetch next A slab
    v16h a = gmem_a_frag(X, KD, mrow, k0, hi);
    v16h b = lds_b_frag(wcol, k0, hi);
    acc = wmma_f16(a, b, acc);
  }

  const float bn = bvec[ncol];
#pragma unroll
  for (int r = 0; r < 8; ++r) {
    int m = mtile * 16 + r + (hi ? 8 : 0);
    size_t idx = (size_t)m * NN + ncol;
    float v = acc[r] + bn;
    if (RES) v += residual[idx];
    Y[idx] = v;
  }
}

// ---------------------------------------------------------------------------
// Kernel 3: per (l, head): S = QK^T/8 + bias; P = softmax(S); ctx = P V.
// Q/K/V^T staged in padded f16 LDS once; softmax writes P back as f16 into the
// (dead) K buffer in A-fragment layout, so both WMMA phases are LDS-fed.
// ctx aliases Q slab in global (block writes exactly the region it read).
// ---------------------------------------------------------------------------
__global__ __launch_bounds__(256)
void gpn_attention(const float* __restrict__ Q, const float* __restrict__ Km,
                   const float* __restrict__ V, const float* __restrict__ biasb,
                   float* __restrict__ ctx) {
  constexpr int P72 = HEAD_DIM + 8;        // padded f16 stride
  __shared__ float    sS[64 * 64];         // 16 KB scores
  __shared__ _Float16 sQ[64 * P72];        // [q][d]
  __shared__ _Float16 sK[64 * P72];        // [key][d]; later reused as P[q][k]
  __shared__ _Float16 sV[64 * P72];        // [d][key]  (transposed V)

  const int bid  = blockIdx.x;
  const int lblk = bid / NUM_HEADS;
  const int h    = bid % NUM_HEADS;
  const size_t slab = (size_t)lblk * 64 * ALL_HEAD + h * HEAD_DIM;
  const float* qg = Q  + slab;
  const float* kg = Km + slab;
  const float* vg = V  + slab;
  float*       cg = ctx + slab;

  // ---- stage Q, K, V^T (float4 loads; 4 vectors per thread) ----
  for (int i = threadIdx.x; i < 1024; i += 256) {
    int row = i >> 4;              // 0..63
    int c4  = (i & 15) << 2;       // 0,4,...,60
    size_t off = (size_t)row * ALL_HEAD + c4;
    float4 q4 = *(const float4*)(qg + off);
    float4 k4 = *(const float4*)(kg + off);
    float4 v4 = *(const float4*)(vg + off);
    _Float16* dq = sQ + row * P72 + c4;
    dq[0] = (_Float16)q4.x; dq[1] = (_Float16)q4.y;
    dq[2] = (_Float16)q4.z; dq[3] = (_Float16)q4.w;
    _Float16* dk = sK + row * P72 + c4;
    dk[0] = (_Float16)k4.x; dk[1] = (_Float16)k4.y;
    dk[2] = (_Float16)k4.z; dk[3] = (_Float16)k4.w;
    sV[(c4 + 0) * P72 + row] = (_Float16)v4.x;
    sV[(c4 + 1) * P72 + row] = (_Float16)v4.y;
    sV[(c4 + 2) * P72 + row] = (_Float16)v4.z;
    sV[(c4 + 3) * P72 + row] = (_Float16)v4.w;
  }
  __syncthreads();

  const int lane = threadIdx.x & 31;
  const int wave = threadIdx.x >> 5;
  const bool hi  = lane >= 16;
  const int ln   = lane & 15;
  const float scale = 0.125f;              // 1/sqrt(64)

  // ---- phase 1: S = QK^T * scale + bias (16 tiles over 8 waves) ----
  for (int t = wave; t < 16; t += 8) {
    int tq = t >> 2, tk = t & 3;
    v8f acc = {};
    for (int k0 = 0; k0 < HEAD_DIM; k0 += 32) {
      v16h a = lds_a_frag(sQ + (tq * 16 + ln) * P72, k0, hi);
      v16h b = lds_b_frag(sK + (tk * 16 + ln) * P72, k0, hi);
      acc = wmma_f16(a, b, acc);
    }
#pragma unroll
    for (int r = 0; r < 8; ++r) {
      int qrow = tq * 16 + r + (hi ? 8 : 0);
      int kcol = tk * 16 + ln;
      sS[qrow * 64 + kcol] = acc[r] * scale + biasb[qrow * 64 + kcol];
    }
  }
  __syncthreads();

  // ---- phase 2: row softmax; emit P as f16 into sK (now dead) ----
  if (threadIdx.x < 64) {
    float* row = sS + threadIdx.x * 64;
    float mx = row[0];
    for (int i = 1; i < 64; ++i) mx = fmaxf(mx, row[i]);
    float sum = 0.0f;
    for (int i = 0; i < 64; ++i) { float e = expf(row[i] - mx); row[i] = e; sum += e; }
    float inv = 1.0f / sum;
    _Float16* prow = sK + threadIdx.x * P72;
    for (int i = 0; i < 64; ++i) prow[i] = (_Float16)(row[i] * inv);
  }
  __syncthreads();

  // ---- phase 3: ctx = P V (P from sK, V^T from sV) ----
  for (int t = wave; t < 16; t += 8) {
    int tq = t >> 2, td = t & 3;
    v8f acc = {};
    for (int k0 = 0; k0 < HEAD_DIM; k0 += 32) {
      v16h a = lds_a_frag(sK + (tq * 16 + ln) * P72, k0, hi);
      v16h b = lds_b_frag(sV + (td * 16 + ln) * P72, k0, hi);
      acc = wmma_f16(a, b, acc);
    }
#pragma unroll
    for (int r = 0; r < 8; ++r) {
      int qrow = tq * 16 + r + (hi ? 8 : 0);
      cg[(size_t)qrow * ALL_HEAD + td * 16 + ln] = acc[r];
    }
  }
}

// ---------------------------------------------------------------------------
// Kernel 4: in-place LayerNorm over H=768 per row. One block per row.
// ---------------------------------------------------------------------------
__global__ __launch_bounds__(256)
void gpn_layernorm(float* __restrict__ X, const float* __restrict__ g,
                   const float* __restrict__ b) {
  __shared__ float red[256];
  float* x = X + (size_t)blockIdx.x * HIDDEN;
  int t = threadIdx.x;
  float v0 = x[t], v1 = x[t + 256], v2 = x[t + 512];

  red[t] = v0 + v1 + v2;
  __syncthreads();
  for (int off = 128; off > 0; off >>= 1) {
    if (t < off) red[t] += red[t + off];
    __syncthreads();
  }
  float mu = red[0] * (1.0f / HIDDEN);
  __syncthreads();

  float d0 = v0 - mu, d1 = v1 - mu, d2 = v2 - mu;
  red[t] = d0 * d0 + d1 * d1 + d2 * d2;
  __syncthreads();
  for (int off = 128; off > 0; off >>= 1) {
    if (t < off) red[t] += red[t + off];
    __syncthreads();
  }
  float inv = rsqrtf(red[0] * (1.0f / HIDDEN) + 1e-12f);

  x[t]       = d0 * inv * g[t]       + b[t];
  x[t + 256] = d1 * inv * g[t + 256] + b[t + 256];
  x[t + 512] = d2 * inv * g[t + 512] + b[t + 512];
}

// ---------------------------------------------------------------------------
extern "C" void kernel_launch(void* const* d_in, const int* in_sizes, int n_in,
                              void* d_out, int out_size, void* d_ws, size_t ws_size,
                              hipStream_t stream) {
  const float* hidden = (const float*)d_in[0];
  const float* source = (const float*)d_in[1];
  const float* amask  = (const float*)d_in[2];
  const float* phylo  = (const float*)d_in[3];
  const float* cc     = (const float*)d_in[4];
  const float* w1     = (const float*)d_in[5];
  const float* w2     = (const float*)d_in[6];
  const float* Wq = (const float*)d_in[7];  const float* bq = (const float*)d_in[8];
  const float* Wk = (const float*)d_in[9];  const float* bk = (const float*)d_in[10];
  const float* Wv = (const float*)d_in[11]; const float* bv = (const float*)d_in[12];
  const float* Wo = (const float*)d_in[13]; const float* bo = (const float*)d_in[14];
  const float* lng = (const float*)d_in[15];
  const float* lnb = (const float*)d_in[16];
  float* out = (float*)d_out;

  // workspace layout (floats): bias[4096] | Q(=ctx) | K | V
  float* ws   = (float*)d_ws;
  float* bias = ws;
  const size_t per = (size_t)SEQ_M * ALL_HEAD;
  float* Qb = ws + 4096;
  float* Kb = Qb + per;
  float* Vb = Kb + per;
  float* Cb = Qb;   // ctx aliases Q (safe: block-local RAW only)

  gpn_fire_bias<<<16, 256, 0, stream>>>(phylo, cc, w1, w2, amask, bias);

  dim3 gQKV(ALL_HEAD / 16, SEQ_M / 128);          // 24 x 512
  gpn_gemm_wmma<HIDDEN, ALL_HEAD, false><<<gQKV, 256, 0, stream>>>(
      hidden, Wq, bq, nullptr, Qb);
  gpn_gemm_wmma<HIDDEN, ALL_HEAD, false><<<gQKV, 256, 0, stream>>>(
      source, Wk, bk, nullptr, Kb);
  gpn_gemm_wmma<HIDDEN, ALL_HEAD, false><<<gQKV, 256, 0, stream>>>(
      source, Wv, bv, nullptr, Vb);

  gpn_attention<<<1024 * NUM_HEADS, 256, 0, stream>>>(Qb, Kb, Vb, bias, Cb);

  dim3 gO(HIDDEN / 16, SEQ_M / 128);              // 48 x 512
  gpn_gemm_wmma<ALL_HEAD, HIDDEN, true><<<gO, 256, 0, stream>>>(
      Cb, Wo, bo, hidden, out);

  gpn_layernorm<<<SEQ_M, 256, 0, stream>>>(out, lng, lnb);
}
